// VectorQuantizer_27221502722181
// MI455X (gfx1250) — compile-verified, matrix operand order checked by reading, not run on hardware
//
#include <hip/hip_runtime.h>
#include <hip/hip_bf16.h>

typedef __attribute__((ext_vector_type(16))) __bf16 v16bf;
typedef __attribute__((ext_vector_type(8)))  __bf16 v8bf;
typedef __attribute__((ext_vector_type(8)))  float  v8f;

#define DB   256      // embedding dim D
#define NK   1024     // number of codes
#define HW   1024     // H*W
#define NTOT 32768    // B*H*W flattened rows
#define ZTOT 8388608  // B*D*H*W
#define LDS_STRIDE 264  // bf16 units per LDS row (256 + 8 pad -> 528B, 16B aligned, conflict-free)
#define CHUNK 32        // codes per LDS chunk (double-buffered)
#define NCHUNK 32       // 1024/32

// ---------------- prep: codebook fp32 -> bf16, row norms, zero loss ----------------
__global__ __launch_bounds__(256) void vq_prep(const float* __restrict__ cb_f,
                                               __bf16* __restrict__ cb_bf,
                                               float* __restrict__ cnorm,
                                               float* __restrict__ loss_acc) {
  const int r = blockIdx.x;           // code row 0..1023
  const int t = threadIdx.x;          // 0..255 = one d each
  float v = cb_f[r * DB + t];
  cb_bf[r * DB + t] = (__bf16)v;
  float s = v * v;
  #pragma unroll
  for (int m = 16; m >= 1; m >>= 1) s += __shfl_xor(s, m, 32);
  __shared__ float red[8];
  if ((t & 31) == 0) red[t >> 5] = s;
  __syncthreads();
  if (t < 8) {
    float x = red[t];
    #pragma unroll
    for (int m = 4; m >= 1; m >>= 1) x += __shfl_xor(x, m, 32);
    if (t == 0) cnorm[r] = x;
  }
  if (r == 0 && t == 0) *loss_acc = 0.0f;   // deterministic re-init every call
}

// ---------------- main: WMMA distance GEMM + fused argmin ----------------
// grid 256 WGs x 256 thr; WG owns 128 rows; wave owns 16 rows; full K=1024 swept via
// double-buffered async global->LDS codebook chunks (32 codes / 16.9KB each).
__global__ __launch_bounds__(256) void vq_main(const float* __restrict__ z,
                                               const __bf16* __restrict__ cb,
                                               const float* __restrict__ cnorm,
                                               int*   __restrict__ idx_i,
                                               float* __restrict__ idx_f) {
  __shared__ __bf16 lds_cb[2][CHUNK * LDS_STRIDE];   // 2 x 16,896 B

  const int tid  = threadIdx.x;
  const int wave = tid >> 5;
  const int lane = tid & 31;
  const int l16  = lane & 15;
  const int hi   = lane >> 4;                 // which K-half this lane holds
  const int rowBase = blockIdx.x * 128 + wave * 16;
  const int n = rowBase + l16;                // this lane's A row (z_flat row)
  // z_flat[n][d] = z[(n/HW)*D*HW + d*HW + (n%HW)]
  const float* zrow = z + (size_t)(n >> 10) * (DB * HW) + (n & (HW - 1));

  // cooperative async-stage mapping: 32 rows x 8 thread-parts x 64B
  const int r8 = tid >> 3;       // row 0..31
  const int p8 = tid & 7;        // 64B part of the row

  // Issue async global->LDS loads for one codebook chunk (per-lane 4 x b128).
  auto issue_chunk = [&](int chunk, int buf) {
    const uint64_t g = (uint64_t)(uintptr_t)(cb + (size_t)(chunk * CHUNK + r8) * DB + p8 * 32);
    const uint32_t l = (uint32_t)(uintptr_t)(&lds_cb[buf][r8 * LDS_STRIDE + p8 * 32]);
    #pragma unroll
    for (int i = 0; i < 4; ++i) {
      asm volatile("global_load_async_to_lds_b128 %0, %1, off"
                   :: "v"((uint32_t)(l + i * 16)), "v"((uint64_t)(g + i * 16))
                   : "memory");
    }
  };

  // Load A panel: 16 rows x 256 d as 8 k-chunks of 16x32 bf16 (ISA A layout:
  // low lanes hold K {0..7,16..23}, high lanes {8..15,24..31} per chunk).
  v16bf A[8];
  #pragma unroll
  for (int kc = 0; kc < 8; ++kc) {
    #pragma unroll
    for (int j = 0; j < 8; ++j) {
      const int d0 = kc * 32 + hi * 8 + j;
      const int d1 = kc * 32 + 16 + hi * 8 + j;
      A[kc][j]     = (__bf16)zrow[d0 * HW];
      A[kc][8 + j] = (__bf16)zrow[d1 * HW];
    }
  }

  float runMin[8];
  int   runIdx[8];
  #pragma unroll
  for (int v = 0; v < 8; ++v) { runMin[v] = 3.4e38f; runIdx[v] = 0; }

  issue_chunk(0, 0);   // prime the pipeline

  for (int chunk = 0; chunk < NCHUNK; ++chunk) {
    asm volatile("s_wait_asynccnt 0x0" ::: "memory");  // own async loads landed
    __syncthreads();                                   // everyone's landed; prev buffer free
    if (chunk + 1 < NCHUNK) issue_chunk(chunk + 1, (chunk + 1) & 1);  // overlap DMA w/ WMMA

    const __bf16* lbuf = lds_cb[chunk & 1];
    #pragma unroll
    for (int tile = 0; tile < 2; ++tile) {    // 16 codes per tile
      v8f acc = {0.f, 0.f, 0.f, 0.f, 0.f, 0.f, 0.f, 0.f};
      const __bf16* brow = lbuf + (tile * 16 + l16) * LDS_STRIDE;
      #pragma unroll
      for (int kc = 0; kc < 8; ++kc) {
        v8bf blo = *(const v8bf*)(brow + kc * 32 + hi * 8);
        v8bf bhi = *(const v8bf*)(brow + kc * 32 + 16 + hi * 8);
        v16bf Bv = __builtin_shufflevector(blo, bhi, 0, 1, 2, 3, 4, 5, 6, 7,
                                                     8, 9, 10, 11, 12, 13, 14, 15);
        acc = __builtin_amdgcn_wmma_f32_16x16x32_bf16(
            /*neg_a=*/false, A[kc], /*neg_b=*/false, Bv,
            /*c_mod=*/(short)0, acc, /*reuse_a=*/false, /*reuse_b=*/false);
      }
      const int code = chunk * CHUNK + tile * 16 + l16;   // this lane's N column
      const float cn = cnorm[code];
      #pragma unroll
      for (int v = 0; v < 8; ++v) {
        const float dist = cn - 2.0f * acc[v];  // ||z||^2 constant per row: dropped
        if (dist < runMin[v]) { runMin[v] = dist; runIdx[v] = code; }  // codes ascend -> first-min kept
      }
    }
  }

  // cross-lane argmin within each 16-lane half (lane = N, vgpr v = row M = v + 8*hi)
  #pragma unroll
  for (int v = 0; v < 8; ++v) {
    float m = runMin[v];
    int ix = runIdx[v];
    #pragma unroll
    for (int s = 8; s >= 1; s >>= 1) {
      float om = __shfl_xor(m, s, 32);
      int   oi = __shfl_xor(ix, s, 32);
      if (om < m || (om == m && oi < ix)) { m = om; ix = oi; }
    }
    if (l16 == 0) {
      const int row = rowBase + v + hi * 8;
      idx_i[row] = ix;
      idx_f[row] = (float)ix;
    }
  }
}

// ---------------- gather z_q + commitment loss partials ----------------
__global__ __launch_bounds__(256) void vq_gather(const float* __restrict__ z,
                                                 const float* __restrict__ cb_f,
                                                 const int* __restrict__ idx_i,
                                                 float* __restrict__ zq,
                                                 float* __restrict__ loss_acc) {
  const int tid = blockIdx.x * 256 + threadIdx.x;   // 0..ZTOT-1
  const int bi = tid >> 18;          // / (D*HW) = /262144
  const int d  = (tid >> 10) & 255;
  const int hw = tid & (HW - 1);
  const int n  = bi * HW + hw;
  const int k  = idx_i[n];
  const float cv = cb_f[k * DB + d];
  const float zv = z[tid];
  zq[tid] = cv;                       // straight-through forward == gathered code
  float df = zv - cv;
  float s = df * df;
  #pragma unroll
  for (int m = 16; m >= 1; m >>= 1) s += __shfl_xor(s, m, 32);
  __shared__ float red[8];
  if ((threadIdx.x & 31) == 0) red[threadIdx.x >> 5] = s;
  __syncthreads();
  if (threadIdx.x == 0) {
    float tot = 0.f;
    #pragma unroll
    for (int i = 0; i < 8; ++i) tot += red[i];
    atomicAdd(loss_acc, tot);
  }
}

__global__ void vq_finalize(const float* __restrict__ loss_acc, float* __restrict__ out_loss) {
  *out_loss = 0.25f * (*loss_acc) / (float)ZTOT;
}

// ---------------- launch ----------------
extern "C" void kernel_launch(void* const* d_in, const int* in_sizes, int n_in,
                              void* d_out, int out_size, void* d_ws, size_t ws_size,
                              hipStream_t stream) {
  const float* z  = (const float*)d_in[0];   // [32,256,32,32]
  const float* cb = (const float*)d_in[1];   // [1024,256]

  float* out   = (float*)d_out;
  float* zq    = out;                         // 8,388,608 f32
  float* idxf  = out + ZTOT;                  // 32,768 indices as float
  float* lossp = out + ZTOT + NTOT;           // 1 scalar

  char* ws = (char*)d_ws;
  __bf16* cb_bf = (__bf16*)ws;                          // 524,288 B
  float* cnorm  = (float*)(ws + 524288);                // 4,096 B
  int*   idxi   = (int*)(ws + 524288 + 4096);           // 131,072 B
  float* lacc   = (float*)(ws + 524288 + 4096 + 131072);

  vq_prep<<<NK, 256, 0, stream>>>(cb, cb_bf, cnorm, lacc);
  vq_main<<<NTOT / 128, 256, 0, stream>>>(z, cb_bf, cnorm, idxi, idxf);
  vq_gather<<<ZTOT / 256, 256, 0, stream>>>(z, cb, idxi, zq, lacc);
  vq_finalize<<<1, 1, 0, stream>>>(lacc, lossp);
}
